// CAM_Module_67010079752269
// MI455X (gfx1250) — matrix operand order repaired and checked by reading, not compiled, not run on hardware
//
#include <hip/hip_runtime.h>
#include <hip/hip_bf16.h>

typedef __attribute__((ext_vector_type(16))) __bf16 v16bf;
typedef __attribute__((ext_vector_type(8)))  __bf16 v8bf;
typedef __attribute__((ext_vector_type(8)))  float  v8f;
typedef __attribute__((ext_vector_type(4)))  float  v4f;

#define NT 128           // columns per block tile
#define NPIX 65536       // H*W
#define CCH 64           // C
#define K3C 192          // 3*C
#define XLD 200          // padded ld for transposed x tile (bf16 elems; 400B rows, 16B aligned)
#define VLD 72           // padded ld for transposed v tile (bf16 elems; 144B rows, 16B aligned)

__device__ inline v8f zero8() {
    v8f z = {0.f, 0.f, 0.f, 0.f, 0.f, 0.f, 0.f, 0.f};
    return z;
}

__device__ inline v8f wmma_bf16(v16bf a, v16bf b, v8f c) {
    return __builtin_amdgcn_wmma_f32_16x16x32_bf16(false, a, false, b, (short)0, c, false, false);
}

// A-fragment (16x32 bf16, MxK) from a row-major [rows x ld] bf16 array.
__device__ inline v16bf load_a_rowmajor(const __bf16* base, int row0, int k0, int ld, int lane) {
    int lr = lane & 15;
    int hi = (lane >> 4) & 1;
    const __bf16* r = base + (long)(row0 + lr) * ld + k0 + hi * 8;
    union { v16bf v; v8bf h[2]; } u;
    u.h[0] = *(const v8bf*)(r);
    u.h[1] = *(const v8bf*)(r + 16);
    return u.v;
}

// B-fragment (32x16 bf16, KxN) from a TRANSPOSED tile: base[col][k] with leading dim ld.
__device__ inline v16bf load_b_tr(const __bf16* base, int col0, int k0, int ld, int lane) {
    const __bf16* r = base + (col0 + (lane & 15)) * ld + k0 + ((lane >> 4) & 1) * 8;
    union { v16bf v; v8bf h[2]; } u;
    u.h[0] = *(const v8bf*)(r);
    u.h[1] = *(const v8bf*)(r + 16);
    return u.v;
}

// ---------------- kernel 0: zero energy accumulator + repack W to frag-ready bf16 ----------------
__global__ __launch_bounds__(256) void prep_kernel(const float* __restrict__ Wq,
                                                   const float* __restrict__ Wk,
                                                   const float* __restrict__ Wv,
                                                   float* __restrict__ energy,
                                                   __bf16* __restrict__ wbf) {
    int idx = blockIdx.x * blockDim.x + threadIdx.x;
    if (idx < 4 * CCH * CCH) energy[idx] = 0.f;   // 16384 floats
    const int total = 3 * CCH * K3C;              // 36864
    for (int t = idx; t < total; t += gridDim.x * blockDim.x) {
        int mat = t / (CCH * K3C);
        int rem = t % (CCH * K3C);
        int row = rem / K3C;
        int k   = rem % K3C;
        const float* W = (mat == 0) ? Wq : (mat == 1) ? Wk : Wv;
        float v = W[row * K3C + k];
        int m  = row >> 4, lr = row & 15;
        int kk = k >> 5,  ko = k & 31;
        int l = lr + (((ko & 15) >= 8) ? 16 : 0);
        int i = (ko & 7) + ((ko >= 16) ? 8 : 0);
        wbf[((((mat * 4 + m) * 6 + kk) * 32) + l) * 16 + i] = (__bf16)v;
    }
}

// ---------------- kernel 1: fused qkv projection + partial energy + v store ----------------
__global__ __launch_bounds__(256) void qkv_energy_kernel(const float* __restrict__ rgb,
                                                         const float* __restrict__ hsv,
                                                         const float* __restrict__ lab,
                                                         const __bf16* __restrict__ wbf,
                                                         const float* __restrict__ bq,
                                                         const float* __restrict__ bk,
                                                         const float* __restrict__ bv,
                                                         __bf16* __restrict__ vbuf,
                                                         float* __restrict__ energy) {
    // transposed x tile: xs[col][ch], ld=XLD -> 51200 B; overlaid later by q/k staging
    __shared__ __bf16 smem[NT * XLD];
    const int b  = blockIdx.y;
    const long n0 = (long)blockIdx.x * NT;
    const int tid  = threadIdx.x;
    const int lane = tid & 31;
    const int wid  = tid >> 5;

    // phase 1: stage x tile transposed as bf16; vectorized f32x4 global reads
    // chunks: 192 channels x 32 col-quads = 6144
    for (int t = tid; t < K3C * (NT / 4); t += 256) {
        int cq = t & 31;        // col quad
        int ch = t >> 5;        // channel 0..191
        const float* s = (ch < 64) ? rgb : (ch < 128) ? hsv : lab;
        v4f v = *(const v4f*)(s + (long)(b * CCH + (ch & 63)) * NPIX + n0 + cq * 4);
#pragma unroll
        for (int j = 0; j < 4; ++j) smem[(cq * 4 + j) * XLD + ch] = (__bf16)v[j];
    }
    __syncthreads();

    // phase 2: each wave computes a 16-column strip of q,k,v (64 rows each)
    const int c0 = wid * 16;
    v8f qa[4], ka[4], va[4];
#pragma unroll
    for (int m = 0; m < 4; ++m) { qa[m] = zero8(); ka[m] = zero8(); va[m] = zero8(); }

#pragma unroll
    for (int kk = 0; kk < 6; ++kk) {
        int k0 = kk * 32;
        v16bf bfrg = load_b_tr(smem, c0, k0, XLD, lane);   // shared across q/k/v and all M tiles
#pragma unroll
        for (int m = 0; m < 4; ++m) {
            v16bf aq = ((const v16bf*)(wbf + (size_t)((0 * 4 + m) * 6 + kk) * 512))[lane];
            v16bf ak = ((const v16bf*)(wbf + (size_t)((1 * 4 + m) * 6 + kk) * 512))[lane];
            v16bf av = ((const v16bf*)(wbf + (size_t)((2 * 4 + m) * 6 + kk) * 512))[lane];
            qa[m] = wmma_bf16(aq, bfrg, qa[m]);
            ka[m] = wmma_bf16(ak, bfrg, ka[m]);
            va[m] = wmma_bf16(av, bfrg, va[m]);
        }
    }

    const int lc = lane & 15;
    const int hi = (lane >> 4) & 1;

    // phase 3: v + bias -> global bf16 v buffer, n-major layout vbuf[b][n][ch]:
    // each lane's 8 accumulator rows are 8 consecutive channels -> one b128 store per m-tile
#pragma unroll
    for (int m = 0; m < 4; ++m) {
        union { v8bf v; __bf16 e[8]; } pv;
#pragma unroll
        for (int r = 0; r < 8; ++r) {
            int row = m * 16 + hi * 8 + r;
            pv.e[r] = (__bf16)(va[m][r] + bv[row]);
        }
        __bf16* dst = vbuf + ((long)b * NPIX + n0 + c0 + lc) * CCH + m * 16 + hi * 8;
        *(v8bf*)dst = pv.v;
    }
    __syncthreads();   // everyone is done reading the x tile

    // phase 4: stage q,k (+bias) bf16 into LDS row-major [64 x NT] (overlaying x tile)
    __bf16* qs = smem;                 // [64 x NT]
    __bf16* ks = smem + CCH * NT;      // [64 x NT]
#pragma unroll
    for (int m = 0; m < 4; ++m) {
#pragma unroll
        for (int r = 0; r < 8; ++r) {
            int row = m * 16 + hi * 8 + r;
            qs[row * NT + c0 + lc] = (__bf16)(qa[m][r] + bq[row]);
            ks[row * NT + c0 + lc] = (__bf16)(ka[m][r] + bk[row]);
        }
    }
    __syncthreads();

    // phase 5: partial energy = q * k^T over this block's NT columns; 16 tiles, 2 per wave
    float* eb = energy + b * CCH * CCH;
#pragma unroll
    for (int t = 0; t < 2; ++t) {
        int tile = wid * 2 + t;
        int cm = tile >> 2, cd = tile & 3;
        v8f e = zero8();
#pragma unroll
        for (int kk = 0; kk < 4; ++kk) {
            v16bf aq  = load_a_rowmajor(qs, cm * 16, kk * 32, NT, lane);
            v16bf bk_ = load_a_rowmajor(ks, cd * 16, kk * 32, NT, lane);
            e = wmma_bf16(aq, bk_, e);
        }
#pragma unroll
        for (int r = 0; r < 8; ++r) {
            int row  = cm * 16 + hi * 8 + r;
            int dcol = cd * 16 + lc;
            atomicAdd(&eb[row * CCH + dcol], e[r]);
        }
    }
}

// ---------------- kernel 2: scaled softmax over energy rows -> bf16 attention ----------------
__global__ __launch_bounds__(256) void softmax_kernel(const float* __restrict__ energy,
                                                      __bf16* __restrict__ attn) {
    int row  = blockIdx.x * (blockDim.x >> 5) + (threadIdx.x >> 5);  // 0..255 (B*C)
    int lane = threadIdx.x & 31;
    const float* e = energy + row * CCH;
    float x0 = e[lane] * 0.125f;          // C^-0.5 = 1/8
    float x1 = e[lane + 32] * 0.125f;
    float mx = fmaxf(x0, x1);
#pragma unroll
    for (int off = 16; off; off >>= 1) mx = fmaxf(mx, __shfl_xor(mx, off, 32));
    float e0 = __expf(x0 - mx), e1 = __expf(x1 - mx);
    float s = e0 + e1;
#pragma unroll
    for (int off = 16; off; off >>= 1) s += __shfl_xor(s, off, 32);
    float inv = 1.f / s;
    attn[row * CCH + lane]      = (__bf16)(e0 * inv);
    attn[row * CCH + lane + 32] = (__bf16)(e1 * inv);
}

// ---------------- kernel 3: out = attention @ v ----------------
__global__ __launch_bounds__(256) void out_kernel(const __bf16* __restrict__ attn,
                                                  const __bf16* __restrict__ vbuf,
                                                  float* __restrict__ out) {
    __shared__ __bf16 vs[NT * VLD];   // transposed v tile: vs[col][ch], padded rows (144 B)
    const int b  = blockIdx.y;
    const long n0 = (long)blockIdx.x * NT;
    const int tid  = threadIdx.x;
    const int lane = tid & 31;
    const int wid  = tid >> 5;

    // stage v tile via CDNA5 async global->LDS copy (ASYNCcnt path).
    // vbuf is n-major: row n = 64 contiguous bf16 = 8 chunks of 16B. 128 rows -> 1024 chunks.
    for (int cc = tid; cc < NT * 8; cc += 256) {
        int col = cc >> 3;
        int seg = cc & 7;
        unsigned lds_off = (unsigned)(unsigned long long)(vs + col * VLD + seg * 8);
        const __bf16* gptr = vbuf + ((long)b * NPIX + n0 + col) * CCH + seg * 8;
        asm volatile("global_load_async_to_lds_b128 %0, %1, off"
                     :: "v"(lds_off), "v"(gptr) : "memory");
    }
    asm volatile("s_wait_asynccnt 0x0" ::: "memory");
    __syncthreads();

    const int c0 = wid * 16;
    const int lc = lane & 15;
    const int hi = (lane >> 4) & 1;
    const __bf16* ab = attn + b * CCH * CCH;

    v8f acc[4];
#pragma unroll
    for (int m = 0; m < 4; ++m) acc[m] = zero8();

#pragma unroll
    for (int kk = 0; kk < 2; ++kk) {
        v16bf bfrg = load_b_tr(vs, c0, kk * 32, VLD, lane);
#pragma unroll
        for (int m = 0; m < 4; ++m) {
            v16bf af = load_a_rowmajor(ab, m * 16, kk * 32, CCH, lane);
            acc[m] = wmma_bf16(af, bfrg, acc[m]);
        }
    }
#pragma unroll
    for (int m = 0; m < 4; ++m) {
#pragma unroll
        for (int r = 0; r < 8; ++r) {
            int row = m * 16 + hi * 8 + r;
            out[(long)(b * CCH + row) * NPIX + n0 + c0 + lc] = acc[m][r];
        }
    }
}

extern "C" void kernel_launch(void* const* d_in, const int* in_sizes, int n_in,
                              void* d_out, int out_size, void* d_ws, size_t ws_size,
                              hipStream_t stream) {
    const float* rgb = (const float*)d_in[0];
    const float* hsv = (const float*)d_in[1];
    const float* lab = (const float*)d_in[2];
    const float* Wq  = (const float*)d_in[3];
    const float* bq  = (const float*)d_in[4];
    const float* Wk  = (const float*)d_in[5];
    const float* bk  = (const float*)d_in[6];
    const float* Wv  = (const float*)d_in[7];
    const float* bv  = (const float*)d_in[8];

    char* ws = (char*)d_ws;
    float*  energy = (float*)ws;                        // 4*64*64*4   = 65536 B
    __bf16* wbf    = (__bf16*)(ws + 65536);             // 3*64*192*2  = 73728 B
    __bf16* attn   = (__bf16*)(ws + 65536 + 73728);     // 4*64*64*2   = 32768 B
    __bf16* vbuf   = (__bf16*)(ws + 172032);            // 4*65536*64*2 = 33.5 MB (n-major)

    prep_kernel<<<144, 256, 0, stream>>>(Wq, Wk, Wv, energy, wbf);

    dim3 gA(NPIX / NT, 4);
    qkv_energy_kernel<<<gA, 256, 0, stream>>>(rgb, hsv, lab, wbf, bq, bk, bv, vbuf, energy);

    softmax_kernel<<<32, 256, 0, stream>>>(energy, attn);

    dim3 gC(NPIX / NT, 4);
    out_kernel<<<gC, 256, 0, stream>>>(attn, vbuf, (float*)d_out);
}
